// QKVAttention_47442208751868
// MI455X (gfx1250) — matrix operand order repaired and checked by reading, not compiled, not run on hardware
//
#include <hip/hip_runtime.h>

// ---------------- problem constants ----------------
#define Bc  2
#define Sc  2048
#define Dc  1024
#define Hc  16
#define HDc 64
#define GM  4096   // B*S
#define GN  1024   // D
#define GK  1024   // D

typedef _Float16 half4 __attribute__((ext_vector_type(4)));
typedef _Float16 half8 __attribute__((ext_vector_type(8)));
typedef _Float16 v16h  __attribute__((ext_vector_type(16)));
typedef float    v8f   __attribute__((ext_vector_type(8)));

#define WMMA_F16(A,B,C) __builtin_amdgcn_wmma_f32_16x16x32_f16(false,(A),false,(B),(short)0,(C),false,false)

static __device__ __forceinline__ half8 ld8(const _Float16* p) { return *(const half8*)p; }
static __device__ __forceinline__ v16h cat8(half8 lo, half8 hi) {
  return __builtin_shufflevector(lo, hi, 0,1,2,3,4,5,6,7,8,9,10,11,12,13,14,15);
}

// LDS byte address of a __shared__ pointer = low 32 bits of the generic address
static __device__ __forceinline__ unsigned lds_addr(const void* p) {
  return (unsigned)(uintptr_t)p;
}
// async DMA: 16 bytes per lane, global -> LDS, tracked by ASYNCcnt
static __device__ __forceinline__ void async_ld16(unsigned lds, const void* gaddr) {
  asm volatile("global_load_async_to_lds_b128 %0, %1, off" :: "v"(lds), "v"(gaddr) : "memory");
}
static __device__ __forceinline__ void wait_async0() {
  asm volatile("s_wait_asynccnt 0x0" ::: "memory");
}

// ---------------- f32 -> f16 convert (4-wide) ----------------
__global__ __launch_bounds__(256) void cvt_f32_f16_x4(const float* __restrict__ s,
                                                      _Float16* __restrict__ d, int n4) {
  int i = blockIdx.x * 256 + threadIdx.x;
  if (i < n4) {
    float4 v = ((const float4*)s)[i];
    half4 o = { (_Float16)v.x, (_Float16)v.y, (_Float16)v.z, (_Float16)v.w };
    ((half4*)d)[i] = o;
  }
}

// ---------------- weight transpose + convert: W[K][N] f32 -> Wt[N][K] f16 ----------------
__global__ __launch_bounds__(256) void transpose_w_f16(const float* __restrict__ W,
                                                       _Float16* __restrict__ Wt) {
  __shared__ _Float16 t[32][33];
  const int bx = blockIdx.x * 32;          // n tile
  const int by = blockIdx.y * 32;          // k tile
  const int tx = threadIdx.x & 31;
  const int ty = threadIdx.x >> 5;         // 0..7
#pragma unroll
  for (int r = 0; r < 32; r += 8)          // read W[k][n], coalesced in n
    t[ty + r][tx] = (_Float16)W[(size_t)(by + ty + r) * GN + bx + tx];
  __syncthreads();
#pragma unroll
  for (int r = 0; r < 32; r += 8)          // write Wt[n][k], coalesced in k
    Wt[(size_t)(bx + ty + r) * GK + by + tx] = t[tx][ty + r];
}

// ---------------- WMMA GEMM: C[M,N] = A[M,K] * Wt[N,K]^T + bias ----------------
// Double-buffered async global->LDS staging; mode 0: f32 row-major, mode 1: f16 [B,H,S,HD]
#define LDA 40                 // 32 + 8 halves pad: conflict-free 16B fragment loads
#define TILE_B (128 * LDA * 2) // bytes per LDS tile buffer
__global__ __launch_bounds__(256) void gemm_f16(const _Float16* __restrict__ A,
                                                const _Float16* __restrict__ Wt,
                                                const float* __restrict__ bias,
                                                float* __restrict__ outF,
                                                _Float16* __restrict__ outH,
                                                int mode) {
  __shared__ _Float16 Asl[2][128 * LDA];   // [m][k] tiles, double buffered
  __shared__ _Float16 Bsl[2][128 * LDA];   // [n][k] tiles, double buffered

  const int tid  = threadIdx.x;
  const int lane = tid & 31;
  const int wave = tid >> 5;
  const int wm   = wave >> 1;              // 0..3 (M)
  const int wn   = wave & 1;               // 0..1 (N)
  const int bm   = blockIdx.x * 128;
  const int bn   = blockIdx.y * 128;
  const int hs   = lane >> 4;
  const int l16  = lane & 15;

  // per-thread 16B copy chunks: two per operand (128x32 halves = 512 chunks)
  const int c0 = tid, c1 = tid + 256;
  const int r0 = c0 >> 2, k80 = (c0 & 3) * 8;
  const int r1 = c1 >> 2, k81 = (c1 & 3) * 8;
  const unsigned aB = lds_addr(&Asl[0][0]);
  const unsigned bB = lds_addr(&Bsl[0][0]);
  const unsigned aOff0 = (unsigned)(r0 * LDA + k80) * 2u;
  const unsigned aOff1 = (unsigned)(r1 * LDA + k81) * 2u;
  const _Float16* gA0 = A  + (size_t)(bm + r0) * GK + k80;
  const _Float16* gA1 = A  + (size_t)(bm + r1) * GK + k81;
  const _Float16* gB0 = Wt + (size_t)(bn + r0) * GK + k80;
  const _Float16* gB1 = Wt + (size_t)(bn + r1) * GK + k81;

  v8f acc[2][4];
#pragma unroll
  for (int mi = 0; mi < 2; ++mi)
#pragma unroll
    for (int ni = 0; ni < 4; ++ni) acc[mi][ni] = (v8f)(0.f);

  auto issue_tile = [&](int kk, int buf) {
    const unsigned o = (unsigned)buf * (unsigned)TILE_B;
    async_ld16(aB + o + aOff0, gA0 + kk);
    async_ld16(aB + o + aOff1, gA1 + kk);
    async_ld16(bB + o + aOff0, gB0 + kk);
    async_ld16(bB + o + aOff1, gB1 + kk);
  };

  issue_tile(0, 0);
  const int NT = GK / 32;                  // 32 iterations
  for (int it = 0; it < NT; ++it) {
    wait_async0();                         // this wave's tile-it copies complete
    __syncthreads();                       // all waves: tile ready, prev compute done
    if (it + 1 < NT) issue_tile((it + 1) * 32, (it + 1) & 1);  // prefetch next tile

    const int buf = it & 1;
    v16h af[2], bf[4];
#pragma unroll
    for (int mi = 0; mi < 2; ++mi) {
      const _Float16* p = &Asl[buf][(wm * 32 + mi * 16 + l16) * LDA + hs * 8];
      af[mi] = cat8(ld8(p), ld8(p + 16));
    }
#pragma unroll
    for (int ni = 0; ni < 4; ++ni) {
      const _Float16* p = &Bsl[buf][(wn * 64 + ni * 16 + l16) * LDA + hs * 8];
      bf[ni] = cat8(ld8(p), ld8(p + 16));
    }
#pragma unroll
    for (int mi = 0; mi < 2; ++mi)
#pragma unroll
      for (int ni = 0; ni < 4; ++ni)
        acc[mi][ni] = WMMA_F16(af[mi], bf[ni], acc[mi][ni]);
  }

  // epilogue: C layout -> VGPR i holds row (i + 8*hs), col l16
#pragma unroll
  for (int mi = 0; mi < 2; ++mi)
#pragma unroll
    for (int ni = 0; ni < 4; ++ni) {
      int col = bn + wn * 64 + ni * 16 + l16;
      float bcol = bias[col];
#pragma unroll
      for (int i = 0; i < 8; ++i) {
        int row = bm + wm * 32 + mi * 16 + i + hs * 8;
        float v = acc[mi][ni][i] + bcol;
        if (mode == 0) {
          outF[(size_t)row * GN + col] = v;
        } else {
          int b = row >> 11, s = row & (Sc - 1);
          int h = col >> 6, hd = col & 63;
          outH[(((size_t)(b * Hc + h) * Sc + s) * HDc) + hd] = (_Float16)v;
        }
      }
    }
}

// ---------------- RoPE in-place on f16 Q/K, [B,H,S,HD]; Q pre-scaled 1/8 ----------------
__global__ __launch_bounds__(256) void rope_qk(_Float16* __restrict__ qh,
                                               _Float16* __restrict__ kh, int n) {
  int i = blockIdx.x * 256 + threadIdx.x;
  if (i >= n) return;
  int hd = i & 31;
  int s = (i >> 5) & (Sc - 1);
  int bh = i >> 16;
  size_t p1 = ((size_t)bh * Sc + s) * HDc + hd;
  size_t p2 = p1 + 32;
  float inv = __expf(-(float)hd * (9.210340372f / 32.f));   // 10000^{-hd/32}
  float ang = (float)s * inv;
  float sn, c;
  __sincosf(ang, &sn, &c);
  float q1 = (float)qh[p1], q2 = (float)qh[p2];
  qh[p1] = (_Float16)((q1 * c - q2 * sn) * 0.125f);
  qh[p2] = (_Float16)((q2 * c + q1 * sn) * 0.125f);
  float k1 = (float)kh[p1], k2 = (float)kh[p2];
  kh[p1] = (_Float16)(k1 * c - k2 * sn);
  kh[p2] = (_Float16)(k2 * c + k1 * sn);
}

// ---------------- flash attention (causal), WMMA, online softmax ----------------
#define LDK 72
#define LDV 40
#define LDP 40
__global__ __launch_bounds__(256) void flash_attn(const _Float16* __restrict__ qh,
                                                  const _Float16* __restrict__ kh,
                                                  const _Float16* __restrict__ vh,
                                                  _Float16* __restrict__ outh) {
  __shared__ _Float16 Ksl[32 * LDK];        // [key][hd]
  __shared__ _Float16 Vsl[64 * LDV];        // [hd][key] (transposed)
  __shared__ _Float16 Psl[8][16 * LDP];     // per-wave P staging 16x32

  const int tid  = threadIdx.x;
  const int lane = tid & 31;
  const int wave = tid >> 5;
  const int hs   = lane >> 4;
  const int l16  = lane & 15;

  const int nQB  = Sc / 128;
  const int qBlk = blockIdx.x % nQB;
  const int bh   = blockIdx.x / nQB;
  const int h    = bh % Hc;
  const int b    = bh / Hc;
  const int qBase = qBlk * 128;
  const int qb    = qBase + wave * 16;

  const _Float16* qp = qh + (size_t)bh * Sc * HDc;
  const _Float16* kp = kh + (size_t)bh * Sc * HDc;
  const _Float16* vp = vh + (size_t)bh * Sc * HDc;

  v16h aq0, aq1;
  {
    const _Float16* p = qp + (size_t)(qb + l16) * HDc + hs * 8;
    aq0 = cat8(ld8(p), ld8(p + 16));
    aq1 = cat8(ld8(p + 32), ld8(p + 48));
  }

  float m[8], ell[8];
#pragma unroll
  for (int i = 0; i < 8; ++i) { m[i] = -1e30f; ell[i] = 0.f; }
  v8f o[4];
#pragma unroll
  for (int t = 0; t < 4; ++t) o[t] = (v8f)(0.f);

  const int krow = tid >> 3;                // 0..31 (key)
  const int k8   = (tid & 7) * 8;           // hd chunk
  const unsigned kDst = lds_addr(&Ksl[0]) + (unsigned)(krow * LDK + k8) * 2u;

  const int kEnd = qBase + 128;
  for (int kb = 0; kb < kEnd; kb += 32) {
    // K tile via async DMA (row-major, contiguous 16B chunks)
    async_ld16(kDst, kp + (size_t)(kb + krow) * HDc + k8);
    // V tile manually transposed into [hd][key]
    {
      half8 w = ld8(vp + (size_t)(kb + krow) * HDc + k8);
#pragma unroll
      for (int j = 0; j < 8; ++j) Vsl[(k8 + j) * LDV + krow] = w[j];
    }
    wait_async0();
    __syncthreads();

    // S = Q K^T
    v8f s0 = (v8f)(0.f), s1 = (v8f)(0.f);
    {
      const _Float16* p0 = &Ksl[l16 * LDK + hs * 8];
      s0 = WMMA_F16(aq0, cat8(ld8(p0), ld8(p0 + 16)), s0);
      s0 = WMMA_F16(aq1, cat8(ld8(p0 + 32), ld8(p0 + 48)), s0);
      const _Float16* p1 = &Ksl[(16 + l16) * LDK + hs * 8];
      s1 = WMMA_F16(aq0, cat8(ld8(p1), ld8(p1 + 16)), s1);
      s1 = WMMA_F16(aq1, cat8(ld8(p1 + 32), ld8(p1 + 48)), s1);
    }

    // causal mask + online softmax
#pragma unroll
    for (int i = 0; i < 8; ++i) {
      int row = qb + i + hs * 8;
      if (kb + l16 > row)      s0[i] = -1e30f;
      if (kb + 16 + l16 > row) s1[i] = -1e30f;
      float mx = fmaxf(s0[i], s1[i]);
#pragma unroll
      for (int d = 8; d >= 1; d >>= 1) mx = fmaxf(mx, __shfl_xor(mx, d, 32));
      float mN = fmaxf(m[i], mx);
      float sf = __expf(m[i] - mN);
      m[i] = mN;
      float p0 = __expf(s0[i] - mN);
      float p1v = __expf(s1[i] - mN);
      float rs = p0 + p1v;
#pragma unroll
      for (int d = 8; d >= 1; d >>= 1) rs += __shfl_xor(rs, d, 32);
      ell[i] = ell[i] * sf + rs;
#pragma unroll
      for (int t = 0; t < 4; ++t) o[t][i] *= sf;
      int prow = i + hs * 8;
      Psl[wave][prow * LDP + l16]      = (_Float16)p0;
      Psl[wave][prow * LDP + 16 + l16] = (_Float16)p1v;
    }
    asm volatile("s_wait_dscnt 0" ::: "memory");   // wave-private LDS RAW

    // O += P V
    {
      const _Float16* pp = &Psl[wave][l16 * LDP + hs * 8];
      v16h ap = cat8(ld8(pp), ld8(pp + 16));
#pragma unroll
      for (int t = 0; t < 4; ++t) {
        const _Float16* pv = &Vsl[(t * 16 + l16) * LDV + hs * 8];
        o[t] = WMMA_F16(ap, cat8(ld8(pv), ld8(pv + 16)), o[t]);
      }
    }
    __syncthreads();
  }

  float rl[8];
#pragma unroll
  for (int i = 0; i < 8; ++i) rl[i] = 1.f / ell[i];
#pragma unroll
  for (int t = 0; t < 4; ++t) {
    int hd = t * 16 + l16;
#pragma unroll
    for (int i = 0; i < 8; ++i) {
      int row = qb + i + hs * 8;
      outh[((size_t)b * Sc + row) * Dc + h * HDc + hd] = (_Float16)(o[t][i] * rl[i]);
    }
  }
}

// ---------------- host launcher ----------------
extern "C" void kernel_launch(void* const* d_in, const int* in_sizes, int n_in,
                              void* d_out, int out_size, void* d_ws, size_t ws_size,
                              hipStream_t stream) {
  const float* x  = (const float*)d_in[0];
  // d_in[1] = additive causal mask: applied analytically, unused
  const float* Wq = (const float*)d_in[2];
  const float* bq = (const float*)d_in[3];
  const float* Wk = (const float*)d_in[4];
  const float* bk = (const float*)d_in[5];
  const float* Wv = (const float*)d_in[6];
  const float* bv = (const float*)d_in[7];
  const float* Wo = (const float*)d_in[8];
  const float* bo = (const float*)d_in[9];
  float* out = (float*)d_out;

  char* ws = (char*)d_ws;
  const size_t MB = 1024 * 1024;
  _Float16* xh  = (_Float16*)(ws + 0);        // 8 MB  [B*S, D] f16
  _Float16* wqt = (_Float16*)(ws + 8 * MB);   // 2 MB  [N,K] f16 (transposed)
  _Float16* wkt = (_Float16*)(ws + 10 * MB);
  _Float16* wvt = (_Float16*)(ws + 12 * MB);
  _Float16* wot = (_Float16*)(ws + 14 * MB);
  _Float16* qh  = (_Float16*)(ws + 16 * MB);  // 8 MB  [B,H,S,HD]
  _Float16* kh  = (_Float16*)(ws + 24 * MB);
  _Float16* vh  = (_Float16*)(ws + 32 * MB);
  _Float16* ah  = (_Float16*)(ws + 40 * MB);  // 8 MB  [B*S, D]

  // 1) activations -> f16; weights -> f16 transposed [N][K]
  cvt_f32_f16_x4<<<(GM * GN / 4) / 256, 256, 0, stream>>>(x, xh, GM * GN / 4);
  dim3 gt(GN / 32, GK / 32);
  transpose_w_f16<<<gt, 256, 0, stream>>>(Wq, wqt);
  transpose_w_f16<<<gt, 256, 0, stream>>>(Wk, wkt);
  transpose_w_f16<<<gt, 256, 0, stream>>>(Wv, wvt);
  transpose_w_f16<<<gt, 256, 0, stream>>>(Wo, wot);

  // 2) Q/K/V projections -> f16 [B,H,S,HD]
  dim3 gg(GM / 128, GN / 128);
  gemm_f16<<<gg, 256, 0, stream>>>(xh, wqt, bq, nullptr, qh, 1);
  gemm_f16<<<gg, 256, 0, stream>>>(xh, wkt, bk, nullptr, kh, 1);
  gemm_f16<<<gg, 256, 0, stream>>>(xh, wvt, bv, nullptr, vh, 1);

  // 3) RoPE in place (Q scaled by 1/sqrt(HD))
  int nr = Bc * Hc * Sc * (HDc / 2);
  rope_qk<<<nr / 256, 256, 0, stream>>>(qh, kh, nr);

  // 4) causal flash attention -> f16 [B,S,D]
  flash_attn<<<Bc * Hc * (Sc / 128), 256, 0, stream>>>(qh, kh, vh, ah);

  // 5) output projection -> f32 d_out
  gemm_f16<<<gg, 256, 0, stream>>>(ah, wot, bo, out, nullptr, 0);
}
